// Block_609885356204
// MI455X (gfx1250) — matrix operand order repaired and checked by reading, not compiled, not run on hardware
//
#include <hip/hip_runtime.h>
#include <hip/hip_bf16.h>
#include <math.h>

#define Bv 2
#define Tv 2048
#define Cv 1024
#define Hv 16
#define HDv 64
#define Ev 8
#define Kv 2
#define FFv 4096
#define BTv (Bv*Tv)   // 4096
#define BHv (Bv*Hv)   // 32

typedef __bf16 bf16;
typedef __attribute__((ext_vector_type(16))) bf16 v16bf;
typedef __attribute__((ext_vector_type(8)))  bf16 v8bf;
typedef __attribute__((ext_vector_type(8)))  float v8f;
typedef __attribute__((ext_vector_type(4)))  int v4i;

union AccU { v8f v; float f[8]; };
union AOpU { v16bf v; v8bf h[2]; bf16 b[16]; };

// ---- async global->LDS staging (CDNA5 ASYNCcnt path), guarded ----
#if __has_builtin(__builtin_amdgcn_global_load_async_to_lds_b128) && \
    __has_builtin(__builtin_amdgcn_s_wait_asynccnt)
#define USE_ASYNC_LDS 1
typedef __attribute__((address_space(1))) v4i gv4i;
typedef __attribute__((address_space(3))) v4i lv4i;
__device__ __forceinline__ void cp16_async(const bf16* g, bf16* l) {
  __builtin_amdgcn_global_load_async_to_lds_b128((gv4i*)(g), (lv4i*)(l), 0, 0);
}
__device__ __forceinline__ void async_wait() {
  __builtin_amdgcn_s_wait_asynccnt(0);
}
#else
__device__ __forceinline__ void cp16_async(const bf16* g, bf16* l) {
  *(v8bf*)l = *(const v8bf*)g;   // fallback: reg copy, still ds_store to LDS
}
__device__ __forceinline__ void async_wait() {}
#endif

// ---- WMMA operand loaders (wave32, per CDNA5 ISA 7.12.2 layouts) ----
__device__ __forceinline__ v16bf load_a_tile(const bf16* base, int lda) {
  int lane = threadIdx.x & 31;
  const bf16* p = base + (lane & 15) * lda + ((lane >> 4) << 3);
  AOpU u;
  u.h[0] = *(const v8bf*)(p);
  u.h[1] = *(const v8bf*)(p + 16);
  return u.v;
}
#define APITCH 264   // 256 + 8 pad (keeps 16B alignment, spreads LDS banks)
__device__ __forceinline__ v16bf load_a_lds(const bf16* Abase, int k8) {
  int lane = threadIdx.x & 31;
  const bf16* p = Abase + (lane & 15) * APITCH + k8 + ((lane >> 4) << 3);
  AOpU u;
  u.h[0] = *(const v8bf*)(p);
  u.h[1] = *(const v8bf*)(p + 16);
  return u.v;
}
__device__ __forceinline__ v16bf load_b_tile(const bf16* base, int ldb) {
  int lane = threadIdx.x & 31;
  const bf16* p = base + (lane & 15) * ldb + ((lane >> 4) << 3);
  AOpU u;
  u.h[0] = *(const v8bf*)(p);
  u.h[1] = *(const v8bf*)(p + 16);
  return u.v;
}
__device__ __forceinline__ v8f wmma_bf(v16bf a, v16bf b, v8f c) {
  return __builtin_amdgcn_wmma_f32_16x16x32_bf16(false, a, false, b, (short)0, c,
                                                 false, false);
}

// ---- small utility kernels ----
__global__ void k_zero_cnt(int* cnt) {
  if (blockIdx.x == 0 && threadIdx.x < Ev) cnt[threadIdx.x] = 0;
}

__global__ void k_convert(const float* in, bf16* out, int total) {
  for (int i = blockIdx.x * blockDim.x + threadIdx.x; i < total;
       i += gridDim.x * blockDim.x)
    out[i] = (bf16)in[i];
}

// out[b][n][k] = in[b][k][n]  (batched transpose + f32->bf16)
__global__ void k_transpose_cvt(const float* in, bf16* out, int batch, int K, int N) {
  int total = batch * K * N;
  for (int i = blockIdx.x * blockDim.x + threadIdx.x; i < total;
       i += gridDim.x * blockDim.x) {
    int b = i / (K * N);
    int rem = i - b * K * N;
    int k = rem / N;
    int n = rem - k * N;
    out[b * K * N + n * K + k] = (bf16)in[i];
  }
}

// ---- QKV projection: per wave, 16 token rows x HD(=64) cols, K=C ----
__global__ void k_qkv(const bf16* __restrict__ xb, const bf16* __restrict__ WT,
                      bf16* __restrict__ out, int transposeOut) {
  int tileId = blockIdx.x * (blockDim.x >> 5) + (threadIdx.x >> 5);
  int lane = threadIdx.x & 31;
  int nlane = lane & 15;
  int mb = (lane >> 4) << 3;
  int bh = tileId >> 7;
  int mt = tileId & 127;
  int b = bh >> 4;
  int h = bh & 15;
  const bf16* A = xb + (b * Tv + mt * 16) * Cv;
  const bf16* Bw = WT + h * HDv * Cv;
  AccU c0{}, c1{}, c2{}, c3{};
  // software-pipelined: preload next k-step while issuing WMMAs
  v16bf a = load_a_tile(A, Cv);
  v16bf b0 = load_b_tile(Bw + 0 * 16 * Cv, Cv);
  v16bf b1 = load_b_tile(Bw + 1 * 16 * Cv, Cv);
  v16bf b2 = load_b_tile(Bw + 2 * 16 * Cv, Cv);
  v16bf b3 = load_b_tile(Bw + 3 * 16 * Cv, Cv);
  for (int kk = 0; kk < Cv; kk += 32) {
    v16bf an = a, bn0 = b0, bn1 = b1, bn2 = b2, bn3 = b3;
    if (kk + 32 < Cv) {
      an = load_a_tile(A + kk + 32, Cv);
      bn0 = load_b_tile(Bw + 0 * 16 * Cv + kk + 32, Cv);
      bn1 = load_b_tile(Bw + 1 * 16 * Cv + kk + 32, Cv);
      bn2 = load_b_tile(Bw + 2 * 16 * Cv + kk + 32, Cv);
      bn3 = load_b_tile(Bw + 3 * 16 * Cv + kk + 32, Cv);
    }
    c0.v = wmma_bf(a, b0, c0.v);
    c1.v = wmma_bf(a, b1, c1.v);
    c2.v = wmma_bf(a, b2, c2.v);
    c3.v = wmma_bf(a, b3, c3.v);
    a = an; b0 = bn0; b1 = bn1; b2 = bn2; b3 = bn3;
  }
#pragma unroll
  for (int r = 0; r < 8; r++) {
    int t = mt * 16 + mb + r;
    if (!transposeOut) {
      bf16* o = out + (bh * Tv + t) * HDv;
      o[0 * 16 + nlane] = (bf16)c0.f[r];
      o[1 * 16 + nlane] = (bf16)c1.f[r];
      o[2 * 16 + nlane] = (bf16)c2.f[r];
      o[3 * 16 + nlane] = (bf16)c3.f[r];
    } else {
      out[(bh * HDv + 0 * 16 + nlane) * Tv + t] = (bf16)c0.f[r];
      out[(bh * HDv + 1 * 16 + nlane) * Tv + t] = (bf16)c1.f[r];
      out[(bh * HDv + 2 * 16 + nlane) * Tv + t] = (bf16)c2.f[r];
      out[(bh * HDv + 3 * 16 + nlane) * Tv + t] = (bf16)c3.f[r];
    }
  }
}

// ---- flash-style causal attention; one wave per 16-row Q tile ----
__global__ void k_attn(const bf16* __restrict__ qb, const bf16* __restrict__ kb,
                       const bf16* __restrict__ vT, bf16* __restrict__ attnb) {
  __shared__ float ps[16][33];
  int tile = blockIdx.x;
  int bh = tile >> 7;
  int mt = tile & 127;
  int b = bh >> 4;
  int h = bh & 15;
  int t0 = mt * 16;
  int lane = threadIdx.x & 31;
  int nlane = lane & 15;
  int mb = (lane >> 4) << 3;
  int koffA = (lane >> 4) << 3;

  const bf16* qbase = qb + (bh * Tv + t0) * HDv;
  v16bf aq0 = load_a_tile(qbase + 0, HDv);
  v16bf aq1 = load_a_tile(qbase + 32, HDv);

  AccU o[4];
  for (int j = 0; j < 4; j++) o[j] = AccU{};
  float rmax[8], rsum[8];
#pragma unroll
  for (int r = 0; r < 8; r++) { rmax[r] = -1e30f; rsum[r] = 0.f; }
  const float scale = 0.03125f;  // C^-0.5

  for (int s0 = 0; s0 < t0 + 16; s0 += 32) {
    AccU sA{}, sB{};
    const bf16* kb0 = kb + (bh * Tv + s0) * HDv;
    const bf16* kb1 = kb + (bh * Tv + s0 + 16) * HDv;
    sA.v = wmma_bf(aq0, load_b_tile(kb0 + 0, HDv), sA.v);
    sA.v = wmma_bf(aq1, load_b_tile(kb0 + 32, HDv), sA.v);
    sB.v = wmma_bf(aq0, load_b_tile(kb1 + 0, HDv), sB.v);
    sB.v = wmma_bf(aq1, load_b_tile(kb1 + 32, HDv), sB.v);

    float cmax[8];
#pragma unroll
    for (int r = 0; r < 8; r++) {
      int trow = t0 + mb + r;
      float vA = (s0 + nlane <= trow) ? sA.f[r] * scale : -1e30f;
      float vB = (s0 + 16 + nlane <= trow) ? sB.f[r] * scale : -1e30f;
      sA.f[r] = vA; sB.f[r] = vB;
      cmax[r] = fmaxf(vA, vB);
    }
#pragma unroll
    for (int off = 1; off < 16; off <<= 1)
#pragma unroll
      for (int r = 0; r < 8; r++)
        cmax[r] = fmaxf(cmax[r], __shfl_xor(cmax[r], off, 32));

    float fac[8], psum[8];
#pragma unroll
    for (int r = 0; r < 8; r++) {
      float nm = fmaxf(rmax[r], cmax[r]);
      fac[r] = expf(rmax[r] - nm);
      rmax[r] = nm;
      float pA = expf(sA.f[r] - nm);
      float pB = expf(sB.f[r] - nm);
      sA.f[r] = pA; sB.f[r] = pB;
      psum[r] = pA + pB;
    }
#pragma unroll
    for (int off = 1; off < 16; off <<= 1)
#pragma unroll
      for (int r = 0; r < 8; r++)
        psum[r] += __shfl_xor(psum[r], off, 32);
#pragma unroll
    for (int r = 0; r < 8; r++) {
      rsum[r] = rsum[r] * fac[r] + psum[r];
      o[0].f[r] *= fac[r]; o[1].f[r] *= fac[r];
      o[2].f[r] *= fac[r]; o[3].f[r] *= fac[r];
    }

    __syncthreads();
#pragma unroll
    for (int r = 0; r < 8; r++) {
      ps[mb + r][nlane] = sA.f[r];
      ps[mb + r][16 + nlane] = sB.f[r];
    }
    __syncthreads();

    AOpU pa;
#pragma unroll
    for (int j = 0; j < 8; j++) {
      pa.b[j] = (bf16)ps[nlane][koffA + j];
      pa.b[8 + j] = (bf16)ps[nlane][16 + koffA + j];
    }
#pragma unroll
    for (int nt = 0; nt < 4; nt++)
      o[nt].v = wmma_bf(pa.v, load_b_tile(vT + (bh * HDv + nt * 16) * Tv + s0, Tv),
                        o[nt].v);
  }

#pragma unroll
  for (int r = 0; r < 8; r++) {
    float inv = 1.f / rsum[r];
    int row = (b * Tv + t0 + mb + r) * Cv + h * HDv;
#pragma unroll
    for (int nt = 0; nt < 4; nt++)
      attnb[row + nt * 16 + nlane] = (bf16)(o[nt].f[r] * inv);
  }
}

// ---- output projection: y = attn @ WoT + bo (f32 out) ----
__global__ void k_proj(const bf16* __restrict__ attnb, const bf16* __restrict__ woT,
                       const float* __restrict__ bo, float* __restrict__ y) {
  int tileId = blockIdx.x * (blockDim.x >> 5) + (threadIdx.x >> 5);
  int lane = threadIdx.x & 31;
  int nlane = lane & 15;
  int mb = (lane >> 4) << 3;
  int mt = tileId >> 4;
  int n0 = (tileId & 15) * 64;
  AccU c[4];
  for (int j = 0; j < 4; j++) c[j] = AccU{};
  const bf16* A = attnb + (mt * 16) * Cv;
  v16bf a = load_a_tile(A, Cv);
  v16bf bt0 = load_b_tile(woT + (n0 + 0) * Cv, Cv);
  v16bf bt1 = load_b_tile(woT + (n0 + 16) * Cv, Cv);
  v16bf bt2 = load_b_tile(woT + (n0 + 32) * Cv, Cv);
  v16bf bt3 = load_b_tile(woT + (n0 + 48) * Cv, Cv);
  for (int kk = 0; kk < Cv; kk += 32) {
    v16bf an = a, bn0 = bt0, bn1 = bt1, bn2 = bt2, bn3 = bt3;
    if (kk + 32 < Cv) {
      an = load_a_tile(A + kk + 32, Cv);
      bn0 = load_b_tile(woT + (n0 + 0) * Cv + kk + 32, Cv);
      bn1 = load_b_tile(woT + (n0 + 16) * Cv + kk + 32, Cv);
      bn2 = load_b_tile(woT + (n0 + 32) * Cv + kk + 32, Cv);
      bn3 = load_b_tile(woT + (n0 + 48) * Cv + kk + 32, Cv);
    }
    c[0].v = wmma_bf(a, bt0, c[0].v);
    c[1].v = wmma_bf(a, bt1, c[1].v);
    c[2].v = wmma_bf(a, bt2, c[2].v);
    c[3].v = wmma_bf(a, bt3, c[3].v);
    a = an; bt0 = bn0; bt1 = bn1; bt2 = bn2; bt3 = bn3;
  }
#pragma unroll
  for (int r = 0; r < 8; r++) {
    int m = mt * 16 + mb + r;
#pragma unroll
    for (int j = 0; j < 4; j++) {
      int col = n0 + j * 16 + nlane;
      y[m * Cv + col] = c[j].f[r] + bo[col];
    }
  }
}

// ---- x1 = x + LN(y); also bf16 copy of x1 ----
__global__ void k_ln_res(const float* __restrict__ x, const float* __restrict__ y,
                         const float* __restrict__ g, const float* __restrict__ bt,
                         float* __restrict__ x1, bf16* __restrict__ x1b) {
  int row = blockIdx.x;
  int tid = threadIdx.x;
  __shared__ float red[256];
  float loc[4];
  float s = 0.f, ss = 0.f;
#pragma unroll
  for (int j = 0; j < 4; j++) {
    int c = tid + j * 256;
    float v = y[row * Cv + c];
    loc[j] = v; s += v; ss += v * v;
  }
  red[tid] = s; __syncthreads();
  for (int o2 = 128; o2 > 0; o2 >>= 1) { if (tid < o2) red[tid] += red[tid + o2]; __syncthreads(); }
  float mean = red[0] / Cv; __syncthreads();
  red[tid] = ss; __syncthreads();
  for (int o2 = 128; o2 > 0; o2 >>= 1) { if (tid < o2) red[tid] += red[tid + o2]; __syncthreads(); }
  float var = red[0] / Cv - mean * mean;
  float rstd = rsqrtf(var + 1e-5f);
#pragma unroll
  for (int j = 0; j < 4; j++) {
    int c = tid + j * 256;
    float v = (loc[j] - mean) * rstd * g[c] + bt[c];
    float o = x[row * Cv + c] + v;
    x1[row * Cv + c] = o;
    x1b[row * Cv + c] = (bf16)o;
  }
}

// ---- noisy top-2 router; one wave per token ----
__global__ void k_router(const float* __restrict__ x1, const float* __restrict__ Wr,
                         const float* __restrict__ br, const float* __restrict__ Wn,
                         const float* __restrict__ bn, const float* __restrict__ noise,
                         int* cnt, int* listTok, int* listSlot, float* listGate) {
  int tok = (blockIdx.x * blockDim.x + threadIdx.x) >> 5;
  int lane = threadIdx.x & 31;
  if (tok >= BTv) return;
  float ar[Ev], an[Ev];
#pragma unroll
  for (int e = 0; e < Ev; e++) { ar[e] = 0.f; an[e] = 0.f; }
  for (int j = 0; j < Cv / 32; j++) {
    int c = lane + j * 32;
    float xv = x1[tok * Cv + c];
#pragma unroll
    for (int e = 0; e < Ev; e++) {
      ar[e] += xv * Wr[c * Ev + e];
      an[e] += xv * Wn[c * Ev + e];
    }
  }
#pragma unroll
  for (int off = 16; off > 0; off >>= 1)
#pragma unroll
    for (int e = 0; e < Ev; e++) {
      ar[e] += __shfl_xor(ar[e], off, 32);
      an[e] += __shfl_xor(an[e], off, 32);
    }
  if (lane == 0) {
    float nv[Ev];
#pragma unroll
    for (int e = 0; e < Ev; e++) {
      float lg = ar[e] + br[e];
      float nl = an[e] + bn[e];
      float sp = (nl > 20.f) ? nl : log1pf(expf(nl));
      nv[e] = lg + noise[tok * Ev + e] * sp;
    }
    int i0 = 0;
    for (int e = 1; e < Ev; e++) if (nv[e] > nv[i0]) i0 = e;
    int i1 = -1;
    for (int e = 0; e < Ev; e++)
      if (e != i0 && (i1 < 0 || nv[e] > nv[i1])) i1 = e;
    float g0 = 1.f / (1.f + expf(nv[i1] - nv[i0]));
    float g1 = 1.f - g0;
    int p0 = atomicAdd(&cnt[i0], 1);
    listTok[i0 * BTv + p0] = tok; listSlot[i0 * BTv + p0] = 0; listGate[i0 * BTv + p0] = g0;
    int p1 = atomicAdd(&cnt[i1], 1);
    listTok[i1 * BTv + p1] = tok; listSlot[i1 * BTv + p1] = 1; listGate[i1 * BTv + p1] = g1;
  }
}

__global__ void k_prefix(const int* cnt, int* offs) {
  if (blockIdx.x == 0 && threadIdx.x == 0) {
    int a = 0;
    for (int e = 0; e < Ev; e++) { offs[e] = a; a += cnt[e]; }
    offs[Ev] = a;
  }
}

// ---- expert GEMM1 + exact GELU, A staged via async LDS double-buffer ----
__global__ void k_ffn1(const bf16* __restrict__ x1b, const bf16* __restrict__ w1T,
                       const float* __restrict__ b1, const int* __restrict__ cnt,
                       const int* __restrict__ offs, const int* __restrict__ listTok,
                       bf16* __restrict__ hbuf) {
  __shared__ bf16 As[2][16][APITCH];
  int bi = blockIdx.x;
  int e = bi >> 12;
  int rem = bi & 4095;
  int mt = rem >> 4;
  int ng = rem & 15;
  int ce = cnt[e];
  if (mt * 16 >= ce) return;
  int w = threadIdx.x >> 5;
  int lane = threadIdx.x & 31;
  int nlane = lane & 15;
  int mb = (lane >> 4) << 3;
  int n0 = ng * 256 + w * 64;
  int oe = offs[e];
  const bf16* Bw = w1T + e * FFv * Cv;

  auto stage = [&](int buf, int kc) {
    for (int i = threadIdx.x; i < 512; i += 128) {   // 16 rows x 32 16B-chunks
      int row = i >> 5;
      int c8 = (i & 31) << 3;
      int li = mt * 16 + row; li = (li < ce) ? li : (ce - 1);
      int tok = listTok[e * BTv + li];
      cp16_async(x1b + tok * Cv + kc + c8, &As[buf][row][c8]);
    }
  };

  AccU c[4];
  for (int j = 0; j < 4; j++) c[j] = AccU{};
  stage(0, 0);
  async_wait();
  __syncthreads();
  const int nch = Cv / 256;   // 4
  for (int ch = 0; ch < nch; ch++) {
    int cur = ch & 1;
    if (ch + 1 < nch) stage(cur ^ 1, (ch + 1) * 256);
#pragma unroll
    for (int k8 = 0; k8 < 256; k8 += 32) {
      int kk = ch * 256 + k8;
      v16bf a = load_a_lds(&As[cur][0][0], k8);
#pragma unroll
      for (int j = 0; j < 4; j++)
        c[j].v = wmma_bf(a, load_b_tile(Bw + (n0 + j * 16) * Cv + kk, Cv), c[j].v);
    }
    async_wait();
    __syncthreads();
  }
#pragma unroll
  for (int r = 0; r < 8; r++) {
    int m = mt * 16 + mb + r;
    if (m < ce) {
      int hrow = oe + m;
#pragma unroll
      for (int j = 0; j < 4; j++) {
        int col = n0 + j * 16 + nlane;
        float v = c[j].f[r] + b1[e * FFv + col];
        float gl = 0.5f * v * (1.0f + erff(v * 0.70710678118654752f));
        hbuf[hrow * FFv + col] = (bf16)gl;
      }
    }
  }
}

// ---- expert GEMM2 + gate scale, scatter to contrib[token][slot][C] ----
__global__ void k_ffn2(const bf16* __restrict__ hbuf, const bf16* __restrict__ w2T,
                       const float* __restrict__ b2, const int* __restrict__ cnt,
                       const int* __restrict__ offs, const int* __restrict__ listTok,
                       const int* __restrict__ listSlot, const float* __restrict__ listGate,
                       float* __restrict__ contrib) {
  __shared__ bf16 As[2][16][APITCH];
  int bi = blockIdx.x;
  int e = bi >> 10;
  int rem = bi & 1023;
  int mt = rem >> 2;
  int ng = rem & 3;
  int ce = cnt[e];
  if (mt * 16 >= ce) return;
  int w = threadIdx.x >> 5;
  int lane = threadIdx.x & 31;
  int nlane = lane & 15;
  int mb = (lane >> 4) << 3;
  int n0 = ng * 256 + w * 64;
  int oe = offs[e];
  const bf16* Bw = w2T + e * Cv * FFv;

  auto stage = [&](int buf, int kc) {
    for (int i = threadIdx.x; i < 512; i += 128) {
      int row = i >> 5;
      int c8 = (i & 31) << 3;
      int li = mt * 16 + row; li = (li < ce) ? li : (ce - 1);
      cp16_async(hbuf + (oe + li) * FFv + kc + c8, &As[buf][row][c8]);
    }
  };

  AccU c[4];
  for (int j = 0; j < 4; j++) c[j] = AccU{};
  stage(0, 0);
  async_wait();
  __syncthreads();
  const int nch = FFv / 256;   // 16
  for (int ch = 0; ch < nch; ch++) {
    int cur = ch & 1;
    if (ch + 1 < nch) stage(cur ^ 1, (ch + 1) * 256);
#pragma unroll
    for (int k8 = 0; k8 < 256; k8 += 32) {
      int kk = ch * 256 + k8;
      v16bf a = load_a_lds(&As[cur][0][0], k8);
#pragma unroll
      for (int j = 0; j < 4; j++)
        c[j].v = wmma_bf(a, load_b_tile(Bw + (n0 + j * 16) * FFv + kk, FFv), c[j].v);
    }
    async_wait();
    __syncthreads();
  }
#pragma unroll
  for (int r = 0; r < 8; r++) {
    int m = mt * 16 + mb + r;
    if (m < ce) {
      int lidx = e * BTv + mt * 16 + m;
      int tok = listTok[lidx];
      int slot = listSlot[lidx];
      float gate = listGate[lidx];
#pragma unroll
      for (int j = 0; j < 4; j++) {
        int col = n0 + j * 16 + nlane;
        contrib[((tok << 1) + slot) * Cv + col] = gate * (c[j].f[r] + b2[e * Cv + col]);
      }
    }
  }
}

// ---- final: out = x1 + LN(contrib[t][0] + contrib[t][1]) ----
__global__ void k_final(const float* __restrict__ x1, const float* __restrict__ contrib,
                        const float* __restrict__ g, const float* __restrict__ bt,
                        float* __restrict__ out) {
  int row = blockIdx.x;
  int tid = threadIdx.x;
  __shared__ float red[256];
  const float* c0 = contrib + (row * 2) * Cv;
  const float* c1 = c0 + Cv;
  float loc[4];
  float s = 0.f, ss = 0.f;
#pragma unroll
  for (int j = 0; j < 4; j++) {
    int c = tid + j * 256;
    float v = c0[c] + c1[c];
    loc[j] = v; s += v; ss += v * v;
  }
  red[tid] = s; __syncthreads();
  for (int o2 = 128; o2 > 0; o2 >>= 1) { if (tid < o2) red[tid] += red[tid + o2]; __syncthreads(); }
  float mean = red[0] / Cv; __syncthreads();
  red[tid] = ss; __syncthreads();
  for (int o2 = 128; o2 > 0; o2 >>= 1) { if (tid < o2) red[tid] += red[tid + o2]; __syncthreads(); }
  float var = red[0] / Cv - mean * mean;
  float rstd = rsqrtf(var + 1e-5f);
#pragma unroll
  for (int j = 0; j < 4; j++) {
    int c = tid + j * 256;
    out[row * Cv + c] = x1[row * Cv + c] + (loc[j] - mean) * rstd * g[c] + bt[c];
  }
}

extern "C" void kernel_launch(void* const* d_in, const int* in_sizes, int n_in,
                              void* d_out, int out_size, void* d_ws, size_t ws_size,
                              hipStream_t stream) {
  const float* x    = (const float*)d_in[0];
  const float* Wq   = (const float*)d_in[1];
  const float* Wk   = (const float*)d_in[2];
  const float* Wv   = (const float*)d_in[3];
  const float* Wo   = (const float*)d_in[4];
  const float* bo   = (const float*)d_in[5];
  const float* ln1g = (const float*)d_in[6];
  const float* ln1b = (const float*)d_in[7];
  const float* Wr   = (const float*)d_in[8];
  const float* br   = (const float*)d_in[9];
  const float* Wn   = (const float*)d_in[10];
  const float* bn   = (const float*)d_in[11];
  const float* W1   = (const float*)d_in[12];
  const float* b1   = (const float*)d_in[13];
  const float* W2   = (const float*)d_in[14];
  const float* b2   = (const float*)d_in[15];
  const float* ln2g = (const float*)d_in[16];
  const float* ln2b = (const float*)d_in[17];
  const float* noise= (const float*)d_in[18];
  float* out = (float*)d_out;

  char* ws = (char*)d_ws;
  size_t off = 0;
  auto alloc = [&](size_t bytes) -> void* {
    void* p = ws + off;
    off = (off + bytes + 255) & ~(size_t)255;
    return p;
  };
  bf16*  xb    = (bf16*)alloc((size_t)BTv * Cv * 2);
  bf16*  wqT   = (bf16*)alloc((size_t)Hv * HDv * Cv * 2);
  bf16*  wkT   = (bf16*)alloc((size_t)Hv * HDv * Cv * 2);
  bf16*  wvT   = (bf16*)alloc((size_t)Hv * HDv * Cv * 2);
  bf16*  woT   = (bf16*)alloc((size_t)Cv * Cv * 2);
  bf16*  w1T   = (bf16*)alloc((size_t)Ev * Cv * FFv * 2);
  bf16*  w2T   = (bf16*)alloc((size_t)Ev * Cv * FFv * 2);
  bf16*  qb    = (bf16*)alloc((size_t)BHv * Tv * HDv * 2);
  bf16*  kbuf  = (bf16*)alloc((size_t)BHv * Tv * HDv * 2);
  bf16*  vT    = (bf16*)alloc((size_t)BHv * Tv * HDv * 2);
  bf16*  attnb = (bf16*)alloc((size_t)BTv * Cv * 2);
  float* y     = (float*)alloc((size_t)BTv * Cv * 4);
  float* x1    = (float*)alloc((size_t)BTv * Cv * 4);
  bf16*  x1b   = (bf16*)alloc((size_t)BTv * Cv * 2);
  int*   cnt   = (int*)alloc(Ev * 4);
  int*   offs  = (int*)alloc((Ev + 1) * 4);
  int*   listTok  = (int*)alloc((size_t)Ev * BTv * 4);
  int*   listSlot = (int*)alloc((size_t)Ev * BTv * 4);
  float* listGate = (float*)alloc((size_t)Ev * BTv * 4);
  bf16*  hbuf  = (bf16*)alloc((size_t)Kv * BTv * FFv * 2);
  float* contrib = (float*)alloc((size_t)BTv * Kv * Cv * 4);

  k_zero_cnt<<<1, 64, 0, stream>>>(cnt);
  k_convert<<<2048, 256, 0, stream>>>(x, xb, BTv * Cv);
  k_transpose_cvt<<<4096, 256, 0, stream>>>(Wq, wqT, Hv, Cv, HDv);
  k_transpose_cvt<<<4096, 256, 0, stream>>>(Wk, wkT, Hv, Cv, HDv);
  k_transpose_cvt<<<4096, 256, 0, stream>>>(Wv, wvT, Hv, Cv, HDv);
  k_transpose_cvt<<<2048, 256, 0, stream>>>(Wo, woT, 1, Cv, Cv);
  k_transpose_cvt<<<8192, 256, 0, stream>>>(W1, w1T, Ev, Cv, FFv);
  k_transpose_cvt<<<8192, 256, 0, stream>>>(W2, w2T, Ev, FFv, Cv);

  k_qkv<<<1024, 128, 0, stream>>>(xb, wqT, qb, 0);
  k_qkv<<<1024, 128, 0, stream>>>(xb, wkT, kbuf, 0);
  k_qkv<<<1024, 128, 0, stream>>>(xb, wvT, vT, 1);

  k_attn<<<BHv * (Tv / 16), 32, 0, stream>>>(qb, kbuf, vT, attnb);
  k_proj<<<1024, 128, 0, stream>>>(attnb, woT, bo, y);
  k_ln_res<<<BTv, 256, 0, stream>>>(x, y, ln1g, ln1b, x1, x1b);

  k_router<<<512, 256, 0, stream>>>(x1, Wr, br, Wn, bn, noise,
                                    cnt, listTok, listSlot, listGate);
  k_prefix<<<1, 1, 0, stream>>>(cnt, offs);

  k_ffn1<<<Ev * 256 * 16, 128, 0, stream>>>(x1b, w1T, b1, cnt, offs, listTok, hbuf);
  k_ffn2<<<Ev * 256 * 4, 128, 0, stream>>>(hbuf, w2T, b2, cnt, offs, listTok,
                                           listSlot, listGate, contrib);
  k_final<<<BTv, 256, 0, stream>>>(x1, contrib, ln2g, ln2b, out);
}
